// MultiHeadAttention_18769007084651
// MI455X (gfx1250) — matrix operand order repaired and checked by reading, not compiled, not run on hardware
//
#include <hip/hip_runtime.h>

typedef __attribute__((ext_vector_type(16))) _Float16 v16h;
typedef __attribute__((ext_vector_type(8)))  _Float16 v8h;
typedef __attribute__((ext_vector_type(8)))  float    v8f;

#define B_   2
#define S_   2048
#define D_   1024
#define H_   16
#define DK_  64
#define HD_  1024
#define NEGINF (-1e9f)

union V16U { v16h v; v8h h[2]; };

// A fragment: 16x32 f16, row-major source. lane l: row = l&15,
// halves[0..15] = A[row][k0..k0+15], k0 = (l<16)?0:16.
__device__ __forceinline__ v16h load_a_frag(const _Float16* base, int ld, int lane) {
  int m  = lane & 15;
  int k0 = (lane < 16) ? 0 : 16;
  const _Float16* p = base + (size_t)m * ld + k0;
  V16U r;
  r.h[0] = *(const v8h*)(p);
  r.h[1] = *(const v8h*)(p + 8);
  return r.v;
}

// B fragment: 32x16 f16 where storage has K contiguous per column:
// base points at (column 0, k-offset 0); lane l: col = l&15, halves = [k0..k0+15], k0 = (l<16)?0:16.
__device__ __forceinline__ v16h load_b_frag(const _Float16* base, int ld, int lane) {
  int n  = lane & 15;
  int k0 = (lane < 16) ? 0 : 16;
  return *(const v16h*)(base + (size_t)n * ld + k0);
}

__device__ __forceinline__ v8f wmma_f16(v16h a, v16h b, v8f c) {
  return __builtin_amdgcn_wmma_f32_16x16x32_f16(false, a, false, b, (short)0, c, false, false);
}

__global__ void cvt_f32_to_f16(const float* __restrict__ in, _Float16* __restrict__ out, int n) {
  int i = blockIdx.x * blockDim.x + threadIdx.x;
  if (i < n) out[i] = (_Float16)in[i];
}

// in[R][C] -> out[C][R] (f32 -> f16)
__global__ void wtrans(const float* __restrict__ in, _Float16* __restrict__ out, int R, int C) {
  int i = blockIdx.x * blockDim.x + threadIdx.x;
  if (i < R * C) {
    int r = i / C, c = i % C;
    out[(size_t)c * R + r] = (_Float16)in[i];
  }
}

// Y = X[4096,1024] @ W (W given transposed: WT[N][K]) + bias, with per-mode epilogue.
// Block tile 128(M) x 64(N), 4 waves, each wave computes 32x64.
// K unrolled by 64 with ping-pong fragment sets: WMMA(set A) -> reload A @ kb+64 ->
// WMMA(set B) -> reload B @ kb+96. No register rotation; steady-state partial waits.
// mode 0: RoPE -> qw16[b][h][s][dk]   mode 1: RoPE -> kw16[b][h][s][dk]
// mode 2: -> vT16[b][h][dv][s]        mode 3: + bias -> out32[row][col] (f32)
__global__ __launch_bounds__(128) void gemm_epi(
    const _Float16* __restrict__ X, const _Float16* __restrict__ WT,
    const float* __restrict__ bias, const float* __restrict__ sin_pos,
    _Float16* __restrict__ out16, float* __restrict__ out32, int mode)
{
  const int K = D_;
  int tid = threadIdx.x, wave = tid >> 5, lane = tid & 31;
  int bm = blockIdx.x, bn = blockIdx.y;
  int rowBase = bm * 128 + wave * 32;

  const _Float16* Xr0 = X + (size_t)rowBase * K;
  const _Float16* Xr1 = X + (size_t)(rowBase + 16) * K;
  const _Float16* Wc  = WT + (size_t)(bn * 64) * K;

  v8f acc[2][4] = {};
  v16h aA[2], bA[4], aB[2], bB[4];
  aA[0] = load_a_frag(Xr0, K, lane);
  aA[1] = load_a_frag(Xr1, K, lane);
#pragma unroll
  for (int t = 0; t < 4; ++t) bA[t] = load_b_frag(Wc + (size_t)(t * 16) * K, K, lane);
  aB[0] = load_a_frag(Xr0 + 32, K, lane);
  aB[1] = load_a_frag(Xr1 + 32, K, lane);
#pragma unroll
  for (int t = 0; t < 4; ++t) bB[t] = load_b_frag(Wc + (size_t)(t * 16) * K + 32, K, lane);

  for (int kb = 0; kb < K; kb += 64) {
    int kA = (kb + 64) & (K - 1);   // branchless wrap; tail reloads k=0 (discarded)
    int kB = (kb + 96) & (K - 1);
    __builtin_prefetch(Xr0 + kA, 0, 1);
#pragma unroll
    for (int i = 0; i < 2; ++i)
#pragma unroll
      for (int t = 0; t < 4; ++t)
        acc[i][t] = wmma_f16(aA[i], bA[t], acc[i][t]);
    aA[0] = load_a_frag(Xr0 + kA, K, lane);
    aA[1] = load_a_frag(Xr1 + kA, K, lane);
#pragma unroll
    for (int t = 0; t < 4; ++t) bA[t] = load_b_frag(Wc + (size_t)(t * 16) * K + kA, K, lane);
#pragma unroll
    for (int i = 0; i < 2; ++i)
#pragma unroll
      for (int t = 0; t < 4; ++t)
        acc[i][t] = wmma_f16(aB[i], bB[t], acc[i][t]);
    aB[0] = load_a_frag(Xr0 + kB, K, lane);
    aB[1] = load_a_frag(Xr1 + kB, K, lane);
#pragma unroll
    for (int t = 0; t < 4; ++t) bB[t] = load_b_frag(Wc + (size_t)(t * 16) * K + kB, K, lane);
  }

  __shared__ float tile[128][65];
  int hi = lane >> 4, ln = lane & 15;
#pragma unroll
  for (int i = 0; i < 2; ++i)
#pragma unroll
    for (int t = 0; t < 4; ++t)
#pragma unroll
      for (int r = 0; r < 8; ++r)
        tile[wave * 32 + i * 16 + r + 8 * hi][t * 16 + ln] = acc[i][t][r];
  __syncthreads();

  int row = tid;                 // 128 rows, one per thread
  int gr = bm * 128 + row;
  int b = gr >> 11, s = gr & (S_ - 1);
#pragma unroll
  for (int c = 0; c < 64; c += 2) {
    int gc = bn * 64 + c;
    float v0 = tile[row][c]     + bias[gc];
    float v1 = tile[row][c + 1] + bias[gc + 1];
    int h = gc >> 6, dk = gc & 63;
    if (mode <= 1) {
      // interleaved RoPE: out[2i] = x[2i]*cos - x[2i+1]*sin ; out[2i+1] = x[2i+1]*cos + x[2i]*sin
      float sn = sin_pos[s * DK_ + dk];
      float cs = sin_pos[s * DK_ + dk + 1];
      float o0 = v0 * cs - v1 * sn;
      float o1 = v1 * cs + v0 * sn;
      size_t base = ((size_t)(b * H_ + h) * S_ + s) * DK_ + dk;
      out16[base]     = (_Float16)o0;
      out16[base + 1] = (_Float16)o1;
    } else if (mode == 2) {
      size_t base = ((size_t)(b * H_ + h) * DK_ + dk) * S_ + s;
      out16[base]      = (_Float16)v0;
      out16[base + S_] = (_Float16)v1;
    } else {
      size_t base = (size_t)gr * D_ + gc;
      out32[base]     = v0;
      out32[base + 1] = v1;
    }
  }
}

// Flash attention: block = 4 waves, each wave owns 16 query rows of one (b,h).
// Per 64-key chunk, issue order: K frags (oldest) -> bias/mask gathers -> score
// WMMAs -> V frags -> softmax VALU (hides V latency) -> P via LDS -> P.V WMMAs.
__global__ __launch_bounds__(128) void flash_attn(
    const _Float16* __restrict__ qw, const _Float16* __restrict__ kw,
    const _Float16* __restrict__ vT, const float* __restrict__ a_bias,
    const int* __restrict__ v_mask, _Float16* __restrict__ out16)
{
  int tid = threadIdx.x, wave = tid >> 5, lane = tid & 31;
  int blk   = blockIdx.x;
  int qtile = blk & 31;       // S_/64 = 32 tiles
  int bh    = blk >> 5;
  int b = bh >> 4, h = bh & 15;
  int qrow0 = qtile * 64 + wave * 16;

  const _Float16* qbase = qw + (size_t)bh * S_ * DK_;
  const _Float16* kbase = kw + (size_t)bh * S_ * DK_;
  const _Float16* vbase = vT + (size_t)bh * DK_ * S_;

  v16h aq0 = load_a_frag(qbase + (size_t)qrow0 * DK_ + 0,  DK_, lane);
  v16h aq1 = load_a_frag(qbase + (size_t)qrow0 * DK_ + 32, DK_, lane);

  int hi = lane >> 4, ln = lane & 15;
  float m_r[8], l_r[8];
  v8f o_acc[4] = {};
#pragma unroll
  for (int r = 0; r < 8; ++r) { m_r[r] = -3e38f; l_r[r] = 0.f; }

  __shared__ __align__(16) _Float16 lds_p[4][16][64];
  _Float16 (*P)[64] = lds_p[wave];

  for (int kb = 0; kb < S_; kb += 64) {
    // ---- K fragments for this chunk: issued first (oldest in flight) ----
    v16h kf[8];
#pragma unroll
    for (int t = 0; t < 4; ++t) {
      kf[2 * t]     = load_b_frag(kbase + (size_t)(kb + t * 16) * DK_ + 0,  DK_, lane);
      kf[2 * t + 1] = load_b_frag(kbase + (size_t)(kb + t * 16) * DK_ + 32, DK_, lane);
    }
    // ---- bias / mask gathers: in flight behind the score WMMAs ----
    float bval[4][8];
    int vm[4];
#pragma unroll
    for (int t = 0; t < 4; ++t) {
      int k = kb + t * 16 + ln;
      vm[t] = v_mask[b * S_ + k];
      const float* bptr = a_bias + (size_t)b * S_ * S_ + k;
#pragma unroll
      for (int r = 0; r < 8; ++r)
        bval[t][r] = bptr[(size_t)(qrow0 + r + 8 * hi) * S_];
    }
    // ---- scores S = Q K^T for a 16x64 slab ----
    v8f sacc[4] = {};
#pragma unroll
    for (int t = 0; t < 4; ++t) {
      sacc[t] = wmma_f16(aq0, kf[2 * t],     sacc[t]);
      sacc[t] = wmma_f16(aq1, kf[2 * t + 1], sacc[t]);
    }
    // ---- V fragments: issued now, consumed after softmax ----
    v16h vf[8];
#pragma unroll
    for (int t = 0; t < 4; ++t) {
      vf[2 * t]     = load_b_frag(vbase + (size_t)(t * 16) * S_ + kb + 0,  S_, lane);
      vf[2 * t + 1] = load_b_frag(vbase + (size_t)(t * 16) * S_ + kb + 32, S_, lane);
    }
    // ---- scale + additive bias + key mask ----
    float sc[4][8];
#pragma unroll
    for (int t = 0; t < 4; ++t)
#pragma unroll
      for (int r = 0; r < 8; ++r) {
        float sv = sacc[t][r] * 0.125f + bval[t][r];
        sc[t][r] = vm[t] ? sv : NEGINF;
      }
    // ---- online softmax (row = r + 8*hi; reduce over the 16-lane column group) ----
    float cmax[8], rsum[8];
#pragma unroll
    for (int r = 0; r < 8; ++r)
      cmax[r] = fmaxf(fmaxf(sc[0][r], sc[1][r]), fmaxf(sc[2][r], sc[3][r]));
#pragma unroll
    for (int off = 1; off < 16; off <<= 1)
#pragma unroll
      for (int r = 0; r < 8; ++r)
        cmax[r] = fmaxf(cmax[r], __shfl_xor(cmax[r], off, 32));
#pragma unroll
    for (int r = 0; r < 8; ++r) {
      float mn    = fmaxf(m_r[r], cmax[r]);
      float scale = __expf(m_r[r] - mn);
      m_r[r] = mn;
      float s4 = 0.f;
#pragma unroll
      for (int t = 0; t < 4; ++t) {
        float p = __expf(sc[t][r] - mn);
        sc[t][r] = p;
        s4 += p;
      }
      rsum[r] = s4;
      l_r[r] *= scale;
#pragma unroll
      for (int t = 0; t < 4; ++t) o_acc[t][r] *= scale;
    }
#pragma unroll
    for (int off = 1; off < 16; off <<= 1)
#pragma unroll
      for (int r = 0; r < 8; ++r)
        rsum[r] += __shfl_xor(rsum[r], off, 32);
#pragma unroll
    for (int r = 0; r < 8; ++r) l_r[r] += rsum[r];

    // ---- probs -> LDS (f16) and reload in A-fragment layout ----
#pragma unroll
    for (int t = 0; t < 4; ++t)
#pragma unroll
      for (int r = 0; r < 8; ++r)
        P[r + 8 * hi][t * 16 + ln] = (_Float16)sc[t][r];

    v16h ap0 = load_a_frag(&P[0][0] + 0,  64, lane);
    v16h ap1 = load_a_frag(&P[0][0] + 32, 64, lane);

    // ---- O += P V  (V fragments landed long ago) ----
#pragma unroll
    for (int t = 0; t < 4; ++t) {
      o_acc[t] = wmma_f16(ap0, vf[2 * t],     o_acc[t]);
      o_acc[t] = wmma_f16(ap1, vf[2 * t + 1], o_acc[t]);
    }
  }

  // ---- normalize and write [b][s][h*DV + dv] as f16 for the output projection ----
#pragma unroll
  for (int t = 0; t < 4; ++t)
#pragma unroll
    for (int r = 0; r < 8; ++r) {
      int j = qrow0 + r + 8 * hi;
      float o = o_acc[t][r] / l_r[r];
      out16[(size_t)(b * S_ + j) * HD_ + h * DK_ + t * 16 + ln] = (_Float16)o;
    }
}

extern "C" void kernel_launch(void* const* d_in, const int* in_sizes, int n_in,
                              void* d_out, int out_size, void* d_ws, size_t ws_size,
                              hipStream_t stream) {
  const float* q      = (const float*)d_in[0];
  const float* k      = (const float*)d_in[1];
  const float* v      = (const float*)d_in[2];
  const float* a_bias = (const float*)d_in[3];
  const float* sinp   = (const float*)d_in[4];
  const int*   vmask  = (const int*)d_in[5];
  const float* Wq = (const float*)d_in[6];
  const float* bq = (const float*)d_in[7];
  const float* Wk = (const float*)d_in[8];
  const float* bk = (const float*)d_in[9];
  const float* Wv = (const float*)d_in[10];
  const float* bv = (const float*)d_in[11];
  const float* Wo = (const float*)d_in[12];
  const float* bo = (const float*)d_in[13];
  float* out = (float*)d_out;

  const size_t XSZ = (size_t)B_ * S_ * D_;   // 4,194,304 halves
  const size_t WSZ = (size_t)D_ * HD_;       // 1,048,576 halves
  _Float16* ws   = (_Float16*)d_ws;
  _Float16* xq16 = ws;
  _Float16* xk16 = xq16 + XSZ;
  _Float16* xv16 = xk16 + XSZ;
  _Float16* wqT  = xv16 + XSZ;
  _Float16* wkT  = wqT + WSZ;
  _Float16* wvT  = wkT + WSZ;
  _Float16* woT  = wvT + WSZ;
  _Float16* qw16 = woT + WSZ;
  _Float16* kw16 = qw16 + XSZ;
  _Float16* vT16 = kw16 + XSZ;
  _Float16* at16 = vT16 + XSZ;

  const int thr = 256;
  cvt_f32_to_f16<<<(int)(XSZ / thr), thr, 0, stream>>>(q, xq16, (int)XSZ);
  cvt_f32_to_f16<<<(int)(XSZ / thr), thr, 0, stream>>>(k, xk16, (int)XSZ);
  cvt_f32_to_f16<<<(int)(XSZ / thr), thr, 0, stream>>>(v, xv16, (int)XSZ);
  wtrans<<<(int)(WSZ / thr), thr, 0, stream>>>(Wq, wqT, D_, HD_);
  wtrans<<<(int)(WSZ / thr), thr, 0, stream>>>(Wk, wkT, D_, HD_);
  wtrans<<<(int)(WSZ / thr), thr, 0, stream>>>(Wv, wvT, D_, HD_);
  wtrans<<<(int)(WSZ / thr), thr, 0, stream>>>(Wo, woT, HD_, D_);

  dim3 ggrid(B_ * S_ / 128, HD_ / 64);
  gemm_epi<<<ggrid, 128, 0, stream>>>(xq16, wqT, bq, sinp, qw16, nullptr, 0);
  gemm_epi<<<ggrid, 128, 0, stream>>>(xk16, wkT, bk, sinp, kw16, nullptr, 1);
  gemm_epi<<<ggrid, 128, 0, stream>>>(xv16, wvT, bv, sinp, vT16, nullptr, 2);

  flash_attn<<<B_ * H_ * (S_ / 64), 128, 0, stream>>>(qw16, kw16, vT16, a_bias, vmask, at16);

  gemm_epi<<<ggrid, 128, 0, stream>>>(at16, woT, bo, nullptr, nullptr, out, 3);
}
